// VQVAE_58265526338134
// MI455X (gfx1250) — matrix operand order repaired and checked by reading, not compiled, not run on hardware
//
#include <hip/hip_runtime.h>
#include <cstdint>
#include <cstddef>

// ---------------------------------------------------------------------------
// VQ-VAE forward for MI455X (gfx1250, wave32).
// All conv / convT / VQ matmuls run through v_wmma_f32_16x16x32_f16.
// B operands (weights / codebook^T) are pre-packed into WMMA fragment order
// so the GEMM hot loop is pure b128 loads + WMMA (4 WMMAs per B fragment).
// ---------------------------------------------------------------------------

typedef _Float16 half_t;
typedef __attribute__((ext_vector_type(16))) _Float16 v16h;
typedef __attribute__((ext_vector_type(8)))  _Float16 v8h;
typedef __attribute__((ext_vector_type(8)))  float    v8f;

#define TPB 256

// Packed-B index: B[k][n] for WMMA 16x16x32 fragments.
// Fragment for (k-tile kt, n-tile nt): lane = (n&15) + 16*((k>>4)&1),
// element j = k&15, stored contiguously per lane (16 halves = one v16h).
__host__ __device__ inline size_t bpack_idx(int k, int n, int K, int N) {
  int nt = n >> 4, kt = k >> 5;
  int lane = (n & 15) + (((k >> 4) & 1) << 4);
  return (((size_t)nt * (K >> 5) + kt) * 32 + lane) * 16 + (k & 15);
}

// ---------------------------- WMMA GEMM ------------------------------------
// C[M][N] (f32) = A[M][K] (f16, row-major) * Bp (f16, fragment-packed)
// Block = 128 threads = 4 waves; each wave owns a 64x16 C tile (4 WMMA acc).
// M % 256 == 0, N % 16 == 0, K % 32 == 0 guaranteed by caller.
__global__ void wmma_gemm_f16(const half_t* __restrict__ A,
                              const half_t* __restrict__ Bp,
                              float* __restrict__ C,
                              int M, int N, int K) {
  const int wave = threadIdx.x >> 5;          // 0..3
  const int lane = threadIdx.x & 31;
  const int hsel = lane >> 4;                 // half-wave selector
  const int l15  = lane & 15;
  const int mbase = (blockIdx.x * 4 + wave) * 64;
  const int nt = blockIdx.y;
  const int bcol = nt * 16 + l15;
  const int ktiles = K >> 5;

  v8f acc0 = {}, acc1 = {}, acc2 = {}, acc3 = {};
  const half_t* bptr = Bp + ((size_t)nt * ktiles * 32 + lane) * 16;
  const half_t* arow = A + (size_t)(mbase + l15) * K + hsel * 16;

  for (int kt = 0; kt < ktiles; ++kt) {
    const v16h b = *(const v16h*)bptr;        // whole B fragment: 2x b128
    bptr += 32 * 16;
    const int kk = kt << 5;
    __builtin_prefetch(bptr, 0, 0);           // global_prefetch_b8
#pragma unroll
    for (int s = 0; s < 4; ++s) {
      // A fragment rows mbase+s*16+l15; lanes 0-15: K=kk+{0..15},
      // lanes 16-31: K=kk+{16..31}.   (matches packed B k order)
      const half_t* ap = arow + (size_t)s * 16 * K + kk;
      v8h alo = *(const v8h*)ap;
      v8h ahi = *(const v8h*)(ap + 8);
      v16h a = __builtin_shufflevector(alo, ahi, 0, 1, 2, 3, 4, 5, 6, 7,
                                       8, 9, 10, 11, 12, 13, 14, 15);
      switch (s) {
        case 0: acc0 = __builtin_amdgcn_wmma_f32_16x16x32_f16(false, a, false, b, (short)0, acc0, false, false); break;
        case 1: acc1 = __builtin_amdgcn_wmma_f32_16x16x32_f16(false, a, false, b, (short)0, acc1, false, false); break;
        case 2: acc2 = __builtin_amdgcn_wmma_f32_16x16x32_f16(false, a, false, b, (short)0, acc2, false, false); break;
        case 3: acc3 = __builtin_amdgcn_wmma_f32_16x16x32_f16(false, a, false, b, (short)0, acc3, false, false); break;
      }
    }
  }
  // C/D layout: VGPR i -> row (tile base) + hsel*8 + i, col = l15.
  {
    float* cp = C + (size_t)(mbase + hsel * 8) * N + bcol;
#pragma unroll
    for (int i = 0; i < 8; ++i) cp[(size_t)i * N] = acc0[i];
    cp += (size_t)16 * N;
#pragma unroll
    for (int i = 0; i < 8; ++i) cp[(size_t)i * N] = acc1[i];
    cp += (size_t)16 * N;
#pragma unroll
    for (int i = 0; i < 8; ++i) cp[(size_t)i * N] = acc2[i];
    cp += (size_t)16 * N;
#pragma unroll
    for (int i = 0; i < 8; ++i) cp[(size_t)i * N] = acc3[i];
  }
}

// ------------------------- layout / pack kernels ---------------------------
__global__ void nchw_to_nhwc(const float* __restrict__ in, float* __restrict__ out,
                             int Bn, int C, int H, int W) {
  size_t i = (size_t)blockIdx.x * blockDim.x + threadIdx.x;
  size_t total = (size_t)Bn * C * H * W;
  if (i >= total) return;
  int w = (int)(i % W); size_t t = i / W;
  int h = (int)(t % H); t /= H;
  int c = (int)(t % C); int b = (int)(t / C);
  out[(((size_t)b * H + h) * W + w) * C + c] = in[i];
}

__global__ void nhwc_to_nchw(const float* __restrict__ in, float* __restrict__ out,
                             int Bn, int C, int H, int W) {
  size_t i = (size_t)blockIdx.x * blockDim.x + threadIdx.x;
  size_t total = (size_t)Bn * C * H * W;
  if (i >= total) return;
  int w = (int)(i % W); size_t t = i / W;
  int h = (int)(t % H); t /= H;
  int c = (int)(t % C); int b = (int)(t / C);
  out[i] = in[(((size_t)b * H + h) * W + w) * C + c];
}

__global__ void f32_to_f16(const float* __restrict__ in, half_t* __restrict__ out,
                           size_t n) {
  size_t i = (size_t)blockIdx.x * blockDim.x + threadIdx.x;
  if (i < n) out[i] = (half_t)in[i];
}

// torch conv weight (Cout,Cin,4,4) f32 -> fragment-packed B over
// [k=cin*16+kh*4+kw][cout], K padded to Kpad.
__global__ void repack_conv_w(const float* __restrict__ w, half_t* __restrict__ Bm,
                              int Cin, int Cout, int Kpad) {
  int i = blockIdx.x * blockDim.x + threadIdx.x;
  int total = Kpad * Cout;
  if (i >= total) return;
  int n = i % Cout, k = i / Cout;
  half_t v = (half_t)0.f;
  if (k < Cin * 16) {
    int kw = k & 3, kh = (k >> 2) & 3, cin = k >> 4;
    v = (half_t)w[(((size_t)n * Cin + cin) * 4 + kh) * 4 + kw];
  }
  Bm[bpack_idx(k, n, Kpad, Cout)] = v;
}

// torch convT weight (Cin,Cout,4,4); parity (eh,ew):
//   oh even -> kh in {1,3}; oh odd -> kh in {0,2}   (same for kw)
__global__ void repack_convT_w(const float* __restrict__ w, half_t* __restrict__ Bm,
                               int Cin, int Cout, int Npad, int eh, int ew) {
  int i = blockIdx.x * blockDim.x + threadIdx.x;
  int K = Cin * 4;
  int total = K * Npad;
  if (i >= total) return;
  int n = i % Npad, k = i / Npad;
  half_t v = (half_t)0.f;
  if (n < Cout) {
    int tw = k & 1, th = (k >> 1) & 1, cin = k >> 2;
    int kh = (eh ? 0 : 1) + 2 * th;
    int kw = (ew ? 0 : 1) + 2 * tw;
    v = (half_t)w[(((size_t)cin * Cout + n) * 4 + kh) * 4 + kw];
  }
  Bm[bpack_idx(k, n, K, Npad)] = v;
}

// codebook (1024,512) f32 -> fragment-packed B over [j(=K dim 512)][n(=1024)]
__global__ void repack_codebook(const float* __restrict__ cb, half_t* __restrict__ Bc) {
  int i = blockIdx.x * blockDim.x + threadIdx.x;
  if (i >= 512 * 1024) return;
  int n = i % 1024, j = i / 1024;
  Bc[bpack_idx(j, n, 512, 1024)] = (half_t)cb[(size_t)n * 512 + j];
}

// ------------------------------ im2col -------------------------------------
// conv k=4 s=2 p=1, input NHWC f32 -> col[M][Kpad] f16, k = cin*16 + kh*4 + kw
__global__ void im2col_conv(const float* __restrict__ in, half_t* __restrict__ col,
                            int Bn, int Hi, int Wi, int Cin,
                            int Ho, int Wo, int Kpad) {
  size_t idx = (size_t)blockIdx.x * blockDim.x + threadIdx.x;
  size_t total = (size_t)Bn * Ho * Wo * Kpad;
  if (idx >= total) return;
  int k = (int)(idx % Kpad);
  size_t m = idx / Kpad;
  half_t v = (half_t)0.f;
  if (k < Cin * 16) {
    int kw = k & 3, kh = (k >> 2) & 3, cin = k >> 4;
    int ow = (int)(m % Wo); size_t t = m / Wo;
    int oh = (int)(t % Ho); int b = (int)(t / Ho);
    int ih = oh * 2 - 1 + kh, iw = ow * 2 - 1 + kw;
    if (ih >= 0 && ih < Hi && iw >= 0 && iw < Wi)
      v = (half_t)in[(((size_t)b * Hi + ih) * Wi + iw) * Cin + cin];
  }
  col[idx] = v;
}

// convT parity-class im2col: K = Cin*4, k = cin*4 + th*2 + tw
__global__ void im2col_convT(const float* __restrict__ in, half_t* __restrict__ col,
                             int Bn, int Hi, int Wi, int Cin,
                             int Hq, int Wq, int eh, int ew) {
  int K = Cin * 4;
  size_t idx = (size_t)blockIdx.x * blockDim.x + threadIdx.x;
  size_t total = (size_t)Bn * Hq * Wq * K;
  if (idx >= total) return;
  int k = (int)(idx % K);
  size_t m = idx / K;
  int tw = k & 1, th = (k >> 1) & 1, cin = k >> 2;
  int yw = (int)(m % Wq); size_t t = m / Wq;
  int yh = (int)(t % Hq); int b = (int)(t / Hq);
  int ih = eh ? (yh + 1 - th) : (yh - th);
  int iw = ew ? (yw + 1 - tw) : (yw - tw);
  half_t v = (half_t)0.f;
  if (ih >= 0 && ih < Hi && iw >= 0 && iw < Wi)
    v = (half_t)in[(((size_t)b * Hi + ih) * Wi + iw) * Cin + cin];
  col[idx] = v;
}

// ------------------------------ batchnorm ----------------------------------
__global__ void bn_stats(const float* __restrict__ x, float* __restrict__ mean,
                         float* __restrict__ var, int M, int C) {
  int c = blockIdx.x;
  float s = 0.f, s2 = 0.f;
  for (int m = threadIdx.x; m < M; m += blockDim.x) {
    float v = x[(size_t)m * C + c];
    s += v; s2 += v * v;
  }
  __shared__ float sh[TPB], sh2[TPB];
  sh[threadIdx.x] = s; sh2[threadIdx.x] = s2;
  __syncthreads();
  for (int o = TPB / 2; o > 0; o >>= 1) {
    if (threadIdx.x < o) { sh[threadIdx.x] += sh[threadIdx.x + o];
                           sh2[threadIdx.x] += sh2[threadIdx.x + o]; }
    __syncthreads();
  }
  if (threadIdx.x == 0) {
    float mu = sh[0] / (float)M;
    mean[c] = mu;
    var[c] = sh2[0] / (float)M - mu * mu;
  }
}

__global__ void bn_relu(float* __restrict__ x, const float* __restrict__ g,
                        const float* __restrict__ b, const float* __restrict__ mean,
                        const float* __restrict__ var, size_t total, int C) {
  size_t i = (size_t)blockIdx.x * blockDim.x + threadIdx.x;
  if (i >= total) return;
  int c = (int)(i % C);
  float v = (x[i] - mean[c]) * rsqrtf(var[c] + 1e-5f) * g[c] + b[c];
  x[i] = v > 0.f ? v : 0.f;
}

// --------------------------- vector quantize -------------------------------
__global__ void row_sumsq(const float* __restrict__ x, float* __restrict__ out, int D) {
  int r = blockIdx.x;
  float s = 0.f;
  for (int j = threadIdx.x; j < D; j += blockDim.x) {
    float v = x[(size_t)r * D + j]; s += v * v;
  }
  __shared__ float sh[TPB];
  sh[threadIdx.x] = s; __syncthreads();
  for (int o = TPB / 2; o > 0; o >>= 1) {
    if (threadIdx.x < o) sh[threadIdx.x] += sh[threadIdx.x + o];
    __syncthreads();
  }
  if (threadIdx.x == 0) out[r] = sh[0];
}

__global__ void vq_argmin(const float* __restrict__ zg, const float* __restrict__ zn,
                          const float* __restrict__ en, int* __restrict__ idx) {
  int r = blockIdx.x;
  float best = 3.4e38f; int bi = 0;
  for (int k = threadIdx.x; k < 1024; k += blockDim.x) {
    float d = zn[r] + en[k] - 2.f * zg[(size_t)r * 1024 + k];
    if (d < best) { best = d; bi = k; }
  }
  __shared__ float sv[TPB]; __shared__ int si[TPB];
  sv[threadIdx.x] = best; si[threadIdx.x] = bi;
  __syncthreads();
  for (int o = TPB / 2; o > 0; o >>= 1) {
    if (threadIdx.x < o) {
      if (sv[threadIdx.x + o] < sv[threadIdx.x] ||
          (sv[threadIdx.x + o] == sv[threadIdx.x] && si[threadIdx.x + o] < si[threadIdx.x])) {
        sv[threadIdx.x] = sv[threadIdx.x + o];
        si[threadIdx.x] = si[threadIdx.x + o];
      }
    }
    __syncthreads();
  }
  if (threadIdx.x == 0) idx[r] = si[0];
}

__global__ void zero1(float* p) { *p = 0.f; }

// gather z_q rows, accumulate ||z_q - z||^2, and emit decoder input as NHWC.
// NCHW flat f = r*512 + j ; z shape [32,512,8,8]
__global__ void vq_gather_loss(const float* __restrict__ cb, const int* __restrict__ idx,
                               const float* __restrict__ z_nchw,
                               float* __restrict__ zq_nhwc, float* __restrict__ lossAcc) {
  int r = blockIdx.x;
  const float* e = cb + (size_t)idx[r] * 512;
  float s = 0.f;
  for (int j = threadIdx.x; j < 512; j += blockDim.x) {
    float zq = e[j];
    size_t f = (size_t)r * 512 + j;
    float dz = zq - z_nchw[f];
    s += dz * dz;
    int w = (int)(f & 7); size_t t = f >> 3;
    int h = (int)(t & 7); t >>= 3;
    int c = (int)(t & 511); int b = (int)(t >> 9);
    zq_nhwc[(((size_t)b * 8 + h) * 8 + w) * 512 + c] = zq;
  }
  __shared__ float sh[TPB];
  sh[threadIdx.x] = s; __syncthreads();
  for (int o = TPB / 2; o > 0; o >>= 1) {
    if (threadIdx.x < o) sh[threadIdx.x] += sh[threadIdx.x + o];
    __syncthreads();
  }
  if (threadIdx.x == 0) atomicAdd(lossAcc, sh[0]);
}

__global__ void write_loss(const float* acc, float* out) {
  // codebook_loss = q_latent + e_latent = 2 * mean((z_q - z)^2)
  *out = 2.f * (*acc) / (2048.f * 512.f);
}

// --------------------------- decoder scatter -------------------------------
__global__ void scatter_parity(const float* __restrict__ pc, float* __restrict__ act,
                               int Bn, int Hq, int Wq, int C, int Npad, int eh, int ew) {
  size_t idx = (size_t)blockIdx.x * blockDim.x + threadIdx.x;
  size_t total = (size_t)Bn * Hq * Wq * C;
  if (idx >= total) return;
  int c = (int)(idx % C); size_t m = idx / C;
  int yw = (int)(m % Wq); size_t t = m / Wq;
  int yh = (int)(t % Hq); int b = (int)(t / Hq);
  int Ho = 2 * Hq, Wo = 2 * Wq;
  int oh = 2 * yh + eh, ow = 2 * yw + ew;
  act[(((size_t)b * Ho + oh) * Wo + ow) * C + c] = pc[m * Npad + c];
}

__global__ void scatter_final(const float* __restrict__ pc, const float* __restrict__ bias,
                              float* __restrict__ out, int Bn, int Hq, int Wq,
                              int eh, int ew) {
  size_t idx = (size_t)blockIdx.x * blockDim.x + threadIdx.x;
  size_t total = (size_t)Bn * Hq * Wq * 3;
  if (idx >= total) return;
  int c = (int)(idx % 3); size_t m = idx / 3;
  int yw = (int)(m % Wq); size_t t = m / Wq;
  int yh = (int)(t % Hq); int b = (int)(t / Hq);
  int Ho = 2 * Hq, Wo = 2 * Wq;
  int oh = 2 * yh + eh, ow = 2 * yw + ew;
  float v = pc[m * 16 + c] + bias[c];
  out[(((size_t)b * 3 + c) * Ho + oh) * Wo + ow] = 1.f / (1.f + expf(-v));
}

// ---------------------------------------------------------------------------
static inline unsigned nb(size_t n) { return (unsigned)((n + TPB - 1) / TPB); }

extern "C" void kernel_launch(void* const* d_in, const int* in_sizes, int n_in,
                              void* d_out, int out_size, void* d_ws, size_t ws_size,
                              hipStream_t stream) {
  (void)in_sizes; (void)n_in; (void)out_size; (void)ws_size;
  // Input order per setup_inputs dict flattening:
  const float* x = (const float*)d_in[0];
  const float* enc_w[5], *enc_g[5], *enc_b[5];
  for (int i = 0; i < 5; ++i) { enc_w[i] = (const float*)d_in[1 + i];
                                enc_g[i] = (const float*)d_in[6 + i];
                                enc_b[i] = (const float*)d_in[11 + i]; }
  const float* cb = (const float*)d_in[16];
  const float* dec_w[5], *dec_g[4], *dec_bt[4];
  for (int i = 0; i < 5; ++i) dec_w[i] = (const float*)d_in[17 + i];
  for (int i = 0; i < 4; ++i) { dec_g[i] = (const float*)d_in[22 + i];
                                dec_bt[i] = (const float*)d_in[26 + i]; }
  const float* dec_bias = (const float*)d_in[30];
  float* out = (float*)d_out;

  // ---- workspace layout (bump allocator, ~630 MB total) ----
  uint8_t* cur = (uint8_t*)d_ws;
  auto alloc = [&](size_t bytes) -> void* {
    void* p = (void*)cur; cur += (bytes + 255) & ~(size_t)255; return p;
  };
  float*  x_nhwc  = (float*)alloc(32ull * 256 * 256 * 3 * 4);
  float*  actA    = (float*)alloc(32ull * 128 * 128 * 64 * 4);   // 134 MB
  float*  actB    = (float*)alloc(32ull * 128 * 128 * 64 * 4);   // 134 MB
  half_t* col     = (half_t*)alloc(131072ull * 1024 * 2);        // 268 MB
  float*  parityC = (float*)alloc(524288ull * 16 * 4);           //  34 MB
  float*  bn_mean = (float*)alloc(512 * 4);
  float*  bn_var  = (float*)alloc(512 * 4);
  float*  z_nchw  = (float*)alloc(2048ull * 512 * 4);
  half_t* z_half  = (half_t*)alloc(2048ull * 512 * 2);
  half_t* cbT     = (half_t*)alloc(512ull * 1024 * 2);
  float*  zg      = (float*)alloc(2048ull * 1024 * 4);
  float*  znorm   = (float*)alloc(2048 * 4);
  float*  enorm   = (float*)alloc(1024 * 4);
  int*    vidx    = (int*)alloc(2048 * 4);
  float*  zq_nhwc = (float*)alloc(2048ull * 512 * 4);
  float*  lossAcc = (float*)alloc(4);

  const int encCin[5]  = {3, 64, 64, 128, 256};
  const int encCout[5] = {64, 64, 128, 256, 512};
  const int encHi[5]   = {256, 128, 64, 32, 16};
  const int encKpad[5] = {64, 1024, 1024, 2048, 4096};
  const int decCin[5]  = {512, 256, 128, 64, 32};
  const int decCout[5] = {256, 128, 64, 32, 3};
  const int decHi[5]   = {8, 16, 32, 64, 128};

  half_t* enc_wb[5];
  for (int l = 0; l < 5; ++l)
    enc_wb[l] = (half_t*)alloc((size_t)encKpad[l] * encCout[l] * 2);
  half_t* dec_wb[5][4];
  for (int l = 0; l < 5; ++l) {
    int Npad = (l == 4) ? 16 : decCout[l];
    for (int p = 0; p < 4; ++p)
      dec_wb[l][p] = (half_t*)alloc((size_t)decCin[l] * 4 * Npad * 2);
  }

  // ---- weight / codebook repacks (cheap, deterministic) ----
  for (int l = 0; l < 5; ++l) {
    size_t n = (size_t)encKpad[l] * encCout[l];
    repack_conv_w<<<nb(n), TPB, 0, stream>>>(enc_w[l], enc_wb[l],
                                             encCin[l], encCout[l], encKpad[l]);
  }
  for (int l = 0; l < 5; ++l) {
    int Npad = (l == 4) ? 16 : decCout[l];
    size_t n = (size_t)decCin[l] * 4 * Npad;
    for (int p = 0; p < 4; ++p)
      repack_convT_w<<<nb(n), TPB, 0, stream>>>(dec_w[l], dec_wb[l][p],
                                                decCin[l], decCout[l], Npad,
                                                p >> 1, p & 1);
  }
  repack_codebook<<<nb(512ull * 1024), TPB, 0, stream>>>(cb, cbT);

  // ---- encoder ----
  nchw_to_nhwc<<<nb(32ull * 3 * 256 * 256), TPB, 0, stream>>>(x, x_nhwc, 32, 3, 256, 256);
  float* acts[2] = {actA, actB};
  const float* cin_ptr = x_nhwc;
  for (int l = 0; l < 5; ++l) {
    int Cin = encCin[l], Cout = encCout[l];
    int Hi = encHi[l], Wi = Hi, Ho = Hi / 2, Wo = Ho;
    int Kpad = encKpad[l];
    size_t M = 32ull * Ho * Wo;
    im2col_conv<<<nb(M * Kpad), TPB, 0, stream>>>(cin_ptr, col, 32, Hi, Wi, Cin,
                                                  Ho, Wo, Kpad);
    float* outp = acts[l & 1];
    wmma_gemm_f16<<<dim3((unsigned)(M / 256), (unsigned)(Cout / 16)), dim3(128), 0,
                    stream>>>(col, enc_wb[l], outp, (int)M, Cout, Kpad);
    bn_stats<<<Cout, TPB, 0, stream>>>(outp, bn_mean, bn_var, (int)M, Cout);
    bn_relu<<<nb(M * Cout), TPB, 0, stream>>>(outp, enc_g[l], enc_b[l],
                                              bn_mean, bn_var, M * Cout, Cout);
    cin_ptr = outp;
  }
  float* z = acts[0];  // layer 4 output -> actA, NHWC [32,8,8,512]

  // ---- vector quantize (rows = NCHW flatten, as in reference .view) ----
  nhwc_to_nchw<<<nb(2048ull * 512), TPB, 0, stream>>>(z, z_nchw, 32, 512, 8, 8);
  f32_to_f16<<<nb(2048ull * 512), TPB, 0, stream>>>(z_nchw, z_half, 2048ull * 512);
  row_sumsq<<<2048, TPB, 0, stream>>>(z_nchw, znorm, 512);
  row_sumsq<<<1024, TPB, 0, stream>>>(cb, enorm, 512);
  wmma_gemm_f16<<<dim3(2048 / 256, 1024 / 16), dim3(128), 0, stream>>>(
      z_half, cbT, zg, 2048, 1024, 512);
  vq_argmin<<<2048, TPB, 0, stream>>>(zg, znorm, enorm, vidx);
  zero1<<<1, 1, 0, stream>>>(lossAcc);
  vq_gather_loss<<<2048, TPB, 0, stream>>>(cb, vidx, z_nchw, zq_nhwc, lossAcc);
  write_loss<<<1, 1, 0, stream>>>(lossAcc, out + 32ull * 3 * 256 * 256);

  // ---- decoder (4 parity-class GEMMs per transposed conv) ----
  const float* din = zq_nhwc;
  for (int l = 0; l < 5; ++l) {
    int Cin = decCin[l], Cout = decCout[l];
    int Npad = (l == 4) ? 16 : Cout;
    int Hi = decHi[l], Wi = Hi, Hq = Hi, Wq = Wi;  // Ho = 2*Hi, parity grid = Hi
    int K = Cin * 4;
    size_t Mq = 32ull * Hq * Wq;
    float* outp = acts[l & 1];
    for (int p = 0; p < 4; ++p) {
      int eh = p >> 1, ew = p & 1;
      im2col_convT<<<nb(Mq * K), TPB, 0, stream>>>(din, col, 32, Hi, Wi, Cin,
                                                   Hq, Wq, eh, ew);
      wmma_gemm_f16<<<dim3((unsigned)(Mq / 256), (unsigned)(Npad / 16)), dim3(128), 0,
                      stream>>>(col, dec_wb[l][p], parityC, (int)Mq, Npad, K);
      if (l < 4)
        scatter_parity<<<nb(Mq * Cout), TPB, 0, stream>>>(parityC, outp, 32, Hq, Wq,
                                                          Cout, Npad, eh, ew);
      else
        scatter_final<<<nb(Mq * 3), TPB, 0, stream>>>(parityC, dec_bias, out,
                                                      32, Hq, Wq, eh, ew);
    }
    if (l < 4) {
      size_t Mo = 32ull * (2 * Hq) * (2 * Wq);
      bn_stats<<<Cout, TPB, 0, stream>>>(outp, bn_mean, bn_var, (int)Mo, Cout);
      bn_relu<<<nb(Mo * Cout), TPB, 0, stream>>>(outp, dec_g[l], dec_bt[l],
                                                 bn_mean, bn_var, Mo * Cout, Cout);
      din = outp;
    }
  }
}
